// CTLNN_29686813950013
// MI455X (gfx1250) — compile-verified, matrix operand order checked by reading, not run on hardware
//
#include <hip/hip_runtime.h>
#include <hip/hip_bf16.h>

// ---------------- problem constants ----------------
#define T_STEPS 1024
#define BATCH   64
#define DIN     256
#define HDIM    512
#define KC      (DIN + HDIM)   // 768  (combined K)
#define N1      (2 * HDIM)     // 1024 (mapped N)
#define KB_C    (KC / 32)      // 24 k-blocks for combined GEMM
#define KB_H    (HDIM / 32)    // 16 k-blocks for H-sized GEMMs
#define NGROUPS 4              // batch tiles of 16 rows
#define WGS_PER_GROUP 16       // N-split workgroups per batch tile
#define THREADS 128            // 4 waves (wave32)
#define DT_C    0.01f
#define LN_EPS_C 1e-5f

typedef __bf16 v16bf __attribute__((ext_vector_type(16)));
typedef float  v8f   __attribute__((ext_vector_type(8)));

// ---------------- workspace layout (bytes) ----------------
#define OFF_SYNC   0u                               // 4KB barrier state
#define OFF_H      4096u                            // 4*16*512 f32   = 128KB
#define OFF_FRAGC  (OFF_H + 131072u)                // 4*16*768 bf16  = 96KB
#define OFF_PART   (OFF_FRAGC + 98304u)             // 4*16*16*2 f32  = 8KB
#define OFF_CORE   (OFF_PART + 8192u)               // 4*16*512 bf16  = 64KB
#define OFF_Z      (OFF_CORE + 65536u)              // 64KB
#define OFF_A1     (OFF_Z + 65536u)                 // 64KB
#define OFF_WM     (OFF_A1 + 65536u)                // 768*1024 bf16  = 1.5MB
#define OFF_WT     (OFF_WM + 1572864u)              // 512*512 bf16
#define OFF_WF1    (OFF_WT + 524288u)
#define OFF_WF2    (OFF_WF1 + 524288u)
#define WS_NEEDED  (OFF_WF2 + 524288u)

// round-to-nearest-even float -> bf16 bits
__device__ __forceinline__ unsigned short f2bf(float f) {
  union { float f; unsigned int u; } v; v.f = f;
  unsigned int r = v.u + 0x7FFFu + ((v.u >> 16) & 1u);
  return (unsigned short)(r >> 16);
}

// Slot of element (row, k) inside a 16xK bf16 WMMA fragment buffer.
// CDNA5 16-bit A/B layout: lane = group*16+row, halfword j within lane,
// k = j + 8*group (j<8) or j + 8 + 8*group (j>=8), per 32-deep k-block.
__device__ __forceinline__ int frag_slot(int row, int k) {
  int kb = k >> 5, kr = k & 31, group, j;
  if (kr < 16) { group = kr >> 3; j = kr & 7; }
  else         { int u = kr - 16; group = u >> 3; j = (u & 7) + 8; }
  return ((kb << 5) + (group << 4) + row) * 16 + j;
}

__device__ __forceinline__ v8f wmma_bf16(v16bf a, v16bf b, v8f c) {
  return __builtin_amdgcn_wmma_f32_16x16x32_bf16(false, a, false, b,
                                                 (short)0, c, false, false);
}

// Issue per-lane async DMA copies global -> LDS (ASYNCcnt tracked; no wait).
// Each lane moves 16B per instruction; THREADS lanes sweep 2KB per pass.
__device__ __forceinline__ void async_issue(const unsigned short* gsrc,
                                            unsigned short* ldst,
                                            int bytes, int tid) {
  unsigned lbase = (unsigned)(size_t)ldst;  // low 32 bits = LDS byte offset
#pragma unroll
  for (int off = 0; off < bytes; off += THREADS * 16) {
    unsigned la = lbase + off + tid * 16;
    const char* gp = (const char*)gsrc + off + tid * 16;
    asm volatile("global_load_async_to_lds_b128 %0, %1, off"
                 :: "v"(la), "v"(gp) : "memory");
  }
}
#define ASYNC_WAIT() asm volatile("s_wait_asynccnt 0x0" ::: "memory")

// Sense-counting barrier across the 16 WGs of one batch group (L2 atomics).
__device__ __forceinline__ void group_barrier(int* cnt, int* gen, int target) {
  __syncthreads();
  __threadfence();
  if (threadIdx.x == 0) {
    int prev = __hip_atomic_fetch_add(cnt, 1, __ATOMIC_ACQ_REL,
                                      __HIP_MEMORY_SCOPE_AGENT);
    if (prev == WGS_PER_GROUP - 1) {
      __hip_atomic_store(cnt, 0, __ATOMIC_RELAXED, __HIP_MEMORY_SCOPE_AGENT);
      __hip_atomic_fetch_add(gen, 1, __ATOMIC_RELEASE,
                             __HIP_MEMORY_SCOPE_AGENT);
    } else {
      while (__hip_atomic_load(gen, __ATOMIC_ACQUIRE,
                               __HIP_MEMORY_SCOPE_AGENT) < target)
        __builtin_amdgcn_s_sleep(1);
    }
  }
  __threadfence();
  __syncthreads();
}

// Pre-shuffle a KxN fp32 weight (row-major) into bf16 WMMA B-fragments:
// dst[((ntile*nKB + kb)*32 + lane)*16 + j]
__global__ void pack_weights(const float* __restrict__ W,
                             unsigned short* __restrict__ dst,
                             int K, int N) {
  int total = K * N;
  int nKB = K >> 5;
  for (int e = blockIdx.x * blockDim.x + threadIdx.x; e < total;
       e += gridDim.x * blockDim.x) {
    int k = e / N, n = e - k * N;
    int nt = n >> 4, nl = n & 15;
    int kb = k >> 5, kr = k & 31, group, j;
    if (kr < 16) { group = kr >> 3; j = kr & 7; }
    else         { int u = kr - 16; group = u >> 3; j = (u & 7) + 8; }
    size_t slot = (((size_t)nt * nKB + kb) * 32 + (group * 16 + nl)) * 16 + j;
    dst[slot] = f2bf(W[e]);
  }
}

__global__ __launch_bounds__(THREADS)
void ctlnn_persistent(const float* __restrict__ x,
                      const unsigned short* __restrict__ WmP,
                      const unsigned short* __restrict__ WtP,
                      const unsigned short* __restrict__ Wf1P,
                      const unsigned short* __restrict__ Wf2P,
                      const float* __restrict__ bm,
                      const float* __restrict__ bf1,
                      const float* __restrict__ bf2,
                      const float* __restrict__ btau,
                      const float* __restrict__ gam,
                      const float* __restrict__ bet,
                      float* __restrict__ h_g,
                      unsigned short* __restrict__ fragCombB,
                      unsigned short* __restrict__ fragCoreB,
                      unsigned short* __restrict__ fragZB,
                      unsigned short* __restrict__ fragA1B,
                      float* __restrict__ part_g,
                      int* __restrict__ sync_g,
                      float* __restrict__ out) {
  const int tid   = threadIdx.x;
  const int wv    = tid >> 5;
  const int lane  = tid & 31;
  const int btile = blockIdx.x >> 4;   // 0..3 batch tile
  const int ns    = blockIdx.x & 15;   // 0..15 N-slice
  const int b0    = btile * 16;
  const int lm    = lane & 15;         // C/D: n within tile
  const int mhi   = (lane >> 4) * 8;   // C/D: row base
  const int wl    = wv & 1;            // wave within pair
  const int isLow = (wv < 2);          // waves 0-1: g/tau/f; waves 2-3: core/a1

  int* cnt = sync_g + btile * 32;
  int* gen = sync_g + btile * 32 + 16;
  int target = 0;

  float*          hrow     = h_g      + (size_t)btile * 16 * HDIM;
  unsigned short* fragComb = fragCombB + (size_t)btile * 16 * KC;
  unsigned short* fragCore = fragCoreB + (size_t)btile * 16 * HDIM;
  unsigned short* fragZ    = fragZB    + (size_t)btile * 16 * HDIM;
  unsigned short* fragA1   = fragA1B   + (size_t)btile * 16 * HDIM;
  float*          part     = part_g    + (size_t)btile * (WGS_PER_GROUP * 16 * 2);

  // LDS staging for A-fragments (filled by async DMA each phase)
  __shared__ __attribute__((aligned(32))) unsigned short ldsComb[16 * KC];   // 24KB
  __shared__ __attribute__((aligned(32))) unsigned short ldsCore[16 * HDIM]; // 16KB
  __shared__ __attribute__((aligned(32))) unsigned short ldsZ[16 * HDIM];    // 16KB
  __shared__ __attribute__((aligned(32))) unsigned short ldsA1[16 * HDIM];   // 16KB
  __shared__ float gbuf[16 * 32];     // g-preactivations (paired cols)
  __shared__ float corebuf[16 * 32];  // core values
  __shared__ float taubuf[16 * 32];
  __shared__ float sbuf[16 * 32];
  __shared__ float murow[16];
  __shared__ float rsrow[16];

  // ---- preload loop-invariant weight fragments into VGPRs ----
  const int nt2    = 2 * ns + wl;                 // 0..31 tile in H-sized GEMMs
  const int ntile1 = (isLow ? 0 : 32) + nt2;      // mapped tile: g-half / core-half
  v16bf bWm[KB_C];  // Wm slice
  v16bf bW3[KB_H];  // Wt (waves 0-1) or Wf1 (waves 2-3)
  v16bf bW4[KB_H];  // Wf2 (used by waves 0-1)
  {
    const unsigned short* bp = WmP + (size_t)ntile1 * (KB_C * 512);
#pragma unroll
    for (int kb = 0; kb < KB_C; ++kb)
      bWm[kb] = *(const v16bf*)(bp + (kb * 32 + lane) * 16);
    const unsigned short* bp3 = (isLow ? WtP : Wf1P) + (size_t)nt2 * (KB_H * 512);
#pragma unroll
    for (int kb = 0; kb < KB_H; ++kb)
      bW3[kb] = *(const v16bf*)(bp3 + (kb * 32 + lane) * 16);
    const unsigned short* bp4 = Wf2P + (size_t)nt2 * (KB_H * 512);
#pragma unroll
    for (int kb = 0; kb < KB_H; ++kb)
      bW4[kb] = *(const v16bf*)(bp4 + (kb * 32 + lane) * 16);
  }

  // ---- pack x_0 slice into combined fragment (h-part is memset-zero) ----
  for (int e = tid; e < 256; e += THREADS) {
    int m = e >> 4, j = e & 15;
    int k = ns * 16 + j;
    float xv = __builtin_nontemporal_load(
        &x[(((size_t)(b0 + m)) * T_STEPS + 0) * DIN + k]);
    fragComb[frag_slot(m, k)] = f2bf(xv);
  }
  ++target; group_barrier(cnt, gen, target);  // x_0 + h_0 visible

  for (int t = 0; t < T_STEPS; ++t) {
    // ---- Phase 1: async-stage combined fragments, then mapped tiles
    async_issue(fragComb, ldsComb, 16 * KC * 2, tid);
    ASYNC_WAIT();
    __syncthreads();
    {
      v8f acc = {0, 0, 0, 0, 0, 0, 0, 0};
#pragma unroll
      for (int kb = 0; kb < KB_C; ++kb) {
        v16bf a = *(const v16bf*)(&ldsComb[(kb * 32 + lane) * 16]);
        acc = wmma_bf16(a, bWm[kb], acc);
      }
      int colG = ntile1 * 16 + lm;   // global mapped column
      float bb = bm[colG];
      float* dstbuf = isLow ? gbuf : corebuf;
#pragma unroll
      for (int r = 0; r < 8; ++r)
        dstbuf[(mhi + r) * 32 + wl * 16 + lm] = acc[r] + bb;
    }
    __syncthreads();
    // owner computes core/z once, scatters bf16 fragments to global
    for (int e = tid; e < 512; e += THREADS) {
      int m = e >> 5, cl = e & 31;
      float gp = gbuf[m * 32 + cl];
      float co = corebuf[m * 32 + cl];
      float g = 1.f / (1.f + expf(-gp));
      float z = g * tanhf(co);
      int slot = frag_slot(m, ns * 32 + cl);
      fragCore[slot] = f2bf(co);
      fragZ[slot]    = f2bf(z);
    }
    ++target; group_barrier(cnt, gen, target);  // core/z fragments complete

    // ---- Phase 3: async-stage core+z; waves 0-1 tau, waves 2-3 a1 (K=512)
    async_issue(fragCore, ldsCore, 16 * HDIM * 2, tid);
    async_issue(fragZ,    ldsZ,    16 * HDIM * 2, tid);
    ASYNC_WAIT();
    __syncthreads();
    if (isLow) {
      v8f acc = {0, 0, 0, 0, 0, 0, 0, 0};
#pragma unroll
      for (int kb = 0; kb < KB_H; ++kb) {
        v16bf a = *(const v16bf*)(&ldsCore[(kb * 32 + lane) * 16]);
        acc = wmma_bf16(a, bW3[kb], acc);
      }
      int col = nt2 * 16 + lm;
      float bb = btau[col];
#pragma unroll
      for (int r = 0; r < 8; ++r) {
        float v = acc[r] + bb;
        float sp = (v > 20.f) ? v : log1pf(expf(v));  // softplus
        taubuf[(mhi + r) * 32 + wl * 16 + lm] = sp + 1e-6f;
      }
    } else {
      v8f acc = {0, 0, 0, 0, 0, 0, 0, 0};
#pragma unroll
      for (int kb = 0; kb < KB_H; ++kb) {
        v16bf a = *(const v16bf*)(&ldsZ[(kb * 32 + lane) * 16]);
        acc = wmma_bf16(a, bW3[kb], acc);
      }
      int col = nt2 * 16 + lm;
      float bb = bf1[col];
#pragma unroll
      for (int r = 0; r < 8; ++r) {
        float v = acc[r] + bb;
        float s = v / (1.f + expf(-v));               // silu
        fragA1[frag_slot(mhi + r, col)] = f2bf(s);    // scatter bf16 fragment
      }
    }
    ++target; group_barrier(cnt, gen, target);  // a1 fragments complete

    // ---- Phase 4: async-stage a1; waves 2-3 pack x_{t+1} while DMA runs;
    //      waves 0-1 f tiles + combine
    async_issue(fragA1, ldsA1, 16 * HDIM * 2, tid);
    if (!isLow && (t + 1 < T_STEPS)) {
      for (int e = tid - 64; e < 256; e += 64) {   // waves 2-3: tid 64..127
        int m = e >> 4, j = e & 15;
        int k = ns * 16 + j;
        float xv = __builtin_nontemporal_load(
            &x[(((size_t)(b0 + m)) * T_STEPS + (t + 1)) * DIN + k]);
        fragComb[frag_slot(m, k)] = f2bf(xv);
      }
    }
    ASYNC_WAIT();
    __syncthreads();
    if (isLow) {
      v8f acc = {0, 0, 0, 0, 0, 0, 0, 0};
#pragma unroll
      for (int kb = 0; kb < KB_H; ++kb) {
        v16bf a = *(const v16bf*)(&ldsA1[(kb * 32 + lane) * 16]);
        acc = wmma_bf16(a, bW4[kb], acc);
      }
      int col = nt2 * 16 + lm;
      float bb = bf2[col];
#pragma unroll
      for (int r = 0; r < 8; ++r) {
        int m = mhi + r;
        float fv = acc[r] + bb;
        float tauv = taubuf[m * 32 + wl * 16 + lm];
        sbuf[m * 32 + wl * 16 + lm] =
            hrow[m * HDIM + col] + (DT_C / tauv) * fv;
      }
    }
    __syncthreads();
    // partial LayerNorm sums over this WG's 32 columns
    if (tid < 16) {
      float sum = 0.f, ss = 0.f;
      for (int c = 0; c < 32; ++c) {
        float s = sbuf[tid * 32 + c];
        sum += s; ss += s * s;
      }
      part[(ns * 16 + tid) * 2 + 0] = sum;
      part[(ns * 16 + tid) * 2 + 1] = ss;
    }
    ++target; group_barrier(cnt, gen, target);  // partials complete

    // ---- Phase 5: LayerNorm, commit h (fp32 + bf16 fragment) and output
    if (tid < 16) {
      float sum = 0.f, ss = 0.f;
      for (int g2 = 0; g2 < WGS_PER_GROUP; ++g2) {
        sum += part[(g2 * 16 + tid) * 2 + 0];
        ss  += part[(g2 * 16 + tid) * 2 + 1];
      }
      float mu = sum * (1.f / HDIM);
      float var = ss * (1.f / HDIM) - mu * mu;
      murow[tid] = mu;
      rsrow[tid] = rsqrtf(var + LN_EPS_C);
    }
    __syncthreads();
    for (int e = tid; e < 512; e += THREADS) {
      int m = e >> 5, cl = e & 31;
      int col = ns * 32 + cl;
      float hn = (sbuf[m * 32 + cl] - murow[m]) * rsrow[m] * gam[col] + bet[col];
      hrow[m * HDIM + col] = hn;
      fragComb[frag_slot(m, DIN + col)] = f2bf(hn);
      __builtin_nontemporal_store(
          hn, &out[(((size_t)(b0 + m)) * T_STEPS + t) * HDIM + col]);
    }
    ++target; group_barrier(cnt, gen, target);  // h committed
  }
}

extern "C" void kernel_launch(void* const* d_in, const int* in_sizes, int n_in,
                              void* d_out, int out_size, void* d_ws,
                              size_t ws_size, hipStream_t stream) {
  (void)in_sizes; (void)n_in; (void)out_size;
  if (ws_size < (size_t)WS_NEEDED) return;

  const float* x    = (const float*)d_in[0];
  const float* Wm   = (const float*)d_in[1];
  const float* bm   = (const float*)d_in[2];
  const float* Wf1  = (const float*)d_in[3];
  const float* bf1  = (const float*)d_in[4];
  const float* Wf2  = (const float*)d_in[5];
  const float* bf2  = (const float*)d_in[6];
  const float* Wt   = (const float*)d_in[7];
  const float* btau = (const float*)d_in[8];
  const float* gam  = (const float*)d_in[9];
  const float* bet  = (const float*)d_in[10];

  char* ws = (char*)d_ws;
  int*            syncp    = (int*)(ws + OFF_SYNC);
  float*          h_g      = (float*)(ws + OFF_H);
  unsigned short* fragComb = (unsigned short*)(ws + OFF_FRAGC);
  float*          part_g   = (float*)(ws + OFF_PART);
  unsigned short* fragCore = (unsigned short*)(ws + OFF_CORE);
  unsigned short* fragZ    = (unsigned short*)(ws + OFF_Z);
  unsigned short* fragA1   = (unsigned short*)(ws + OFF_A1);
  unsigned short* WmP      = (unsigned short*)(ws + OFF_WM);
  unsigned short* WtP      = (unsigned short*)(ws + OFF_WT);
  unsigned short* Wf1P     = (unsigned short*)(ws + OFF_WF1);
  unsigned short* Wf2P     = (unsigned short*)(ws + OFF_WF2);

  // zero barrier state + h0 (fp32 and bf16 fragment h-part) every launch
  hipMemsetAsync(ws, 0, OFF_PART, stream);

  pack_weights<<<512, 256, 0, stream>>>(Wm, WmP, KC, N1);
  pack_weights<<<256, 256, 0, stream>>>(Wt, WtP, HDIM, HDIM);
  pack_weights<<<256, 256, 0, stream>>>(Wf1, Wf1P, HDIM, HDIM);
  pack_weights<<<256, 256, 0, stream>>>(Wf2, Wf2P, HDIM, HDIM);

  ctlnn_persistent<<<NGROUPS * WGS_PER_GROUP, THREADS, 0, stream>>>(
      x, WmP, WtP, Wf1P, Wf2P, bm, bf1, bf2, btau, gam, bet,
      h_g, fragComb, fragCore, fragZ, fragA1, part_g, syncp, (float*)d_out);
}
